// GraphNetwork_43482248905016
// MI455X (gfx1250) — matrix operand order repaired and checked by reading, not compiled, not run on hardware
//
#include <hip/hip_runtime.h>

// ---------------- problem constants ----------------
#define NNODES 65536
#define DDIM   128       // D
#define TD     384       // 3*D
#define DEG    16
#define NHEAD  4
#define HD     32        // head dim == wave32

typedef __bf16 v16bf __attribute__((ext_vector_type(16)));
typedef float  v8f   __attribute__((ext_vector_type(8)));

__device__ __forceinline__ unsigned short f2bf(float f) {
    union { float f; unsigned u; } x; x.f = f;
    unsigned r = x.u + 0x7FFFu + ((x.u >> 16) & 1u);   // round to nearest even
    return (unsigned short)(r >> 16);
}

// -------- WMMA fragment loads (ISA 7.12.2 layouts, bf16, wave32) --------
// A: 16x32 (MxK). lanes 0-15: row M=lane, K chunks [k0..k0+7] & [k0+16..k0+23]
//                 lanes 16-31: row M=lane-16, K chunks [k0+8..15] & [k0+24..31]
__device__ __forceinline__ v16bf load_frag_a(const unsigned short* A, int lda,
                                             int m0, int k0, int lane) {
    int half = lane >> 4, r = lane & 15;
    const unsigned short* p = A + (size_t)(m0 + r) * lda + k0 + half * 8;
    union { v16bf v; uint4 u[2]; } f;
    f.u[0] = *(const uint4*)(p);        // K = k0+half*8 .. +7
    f.u[1] = *(const uint4*)(p + 16);   // K = k0+16+half*8 .. +7
    return f.v;
}
// B: 32x16 (KxN), B[k][n] = W[n][k] (W row-major [Ncols x 128]).
// lanes 0-15: col N=lane, K chunks [k0..k0+7] & [k0+16..k0+23] ; lanes 16-31: col N=lane-16, K chunks [k0+8..15] & [k0+24..31]
__device__ __forceinline__ v16bf load_frag_b(const unsigned short* W, int ldw,
                                             int n0, int k0, int lane) {
    int col = lane & 15, koff = (lane >> 4) * 8;
    const unsigned short* p = W + (size_t)(n0 + col) * ldw + k0 + koff;
    union { v16bf v; uint4 u[2]; } f;
    f.u[0] = *(const uint4*)(p);
    f.u[1] = *(const uint4*)(p + 16);
    return f.v;
}

#define WMMA_BF16(a, b, c) \
    __builtin_amdgcn_wmma_f32_16x16x32_bf16(false, (a), false, (b), (short)0, (c), false, false)

// ---------------- fp32 -> bf16 conversion ----------------
__global__ void cvt_f32_bf16_kernel(const float* __restrict__ src,
                                    unsigned short* __restrict__ dst, int n) {
    int i = blockIdx.x * blockDim.x + threadIdx.x;
    if (i < n) dst[i] = f2bf(src[i]);
}

// ---------------- fused GRU: h' = GRU(x, h) ----------------
// 8 waves/block; block owns 16 rows x all 384 cols of gi & gh.
__global__ __launch_bounds__(256) void gru_fused_kernel(
    const unsigned short* __restrict__ Xb,   // [N,128] bf16 (gate input)
    const unsigned short* __restrict__ Hb,   // [N,128] bf16 (state, bf16)
    const unsigned short* __restrict__ Wih,  // [384,128] bf16
    const unsigned short* __restrict__ Whh,  // [384,128] bf16
    const float* __restrict__ b_ih, const float* __restrict__ b_hh,
    const float* __restrict__ Hprev,         // [N,128] f32 (state, f32)
    float* __restrict__ Hout,                // [N,128] f32
    unsigned short* __restrict__ HoutB,      // [N,128] bf16 (nullable)
    const unsigned char* __restrict__ mask,  // [N,DEG] (nullable)
    int use_mask)
{
    __shared__ float gi[16][TD];
    __shared__ float gh[16][TD];
    __shared__ int   hasnbr[16];

    const int m0   = blockIdx.x * 16;
    const int lane = threadIdx.x & 31;
    const int wave = threadIdx.x >> 5;

    // --- GEMM phase: wave handles col tiles {wave, wave+8, wave+16} ---
    v8f ci[3], cg[3];
#pragma unroll
    for (int t = 0; t < 3; ++t) { ci[t] = (v8f){}; cg[t] = (v8f){}; }

#pragma unroll
    for (int k0 = 0; k0 < DDIM; k0 += 32) {
        v16bf ax = load_frag_a(Xb, DDIM, m0, k0, lane);
        v16bf ah = load_frag_a(Hb, DDIM, m0, k0, lane);
        __builtin_prefetch(Wih + (size_t)(wave * 16) * DDIM + k0 + 32, 0, 1);
        __builtin_prefetch(Whh + (size_t)(wave * 16) * DDIM + k0 + 32, 0, 1);
#pragma unroll
        for (int t = 0; t < 3; ++t) {
            int n0 = (wave + t * 8) * 16;
            v16bf bi = load_frag_b(Wih, DDIM, n0, k0, lane);
            v16bf bh = load_frag_b(Whh, DDIM, n0, k0, lane);
            ci[t] = WMMA_BF16(ax, bi, ci[t]);
            cg[t] = WMMA_BF16(ah, bh, cg[t]);
        }
    }
    {
        const int half = lane >> 4, col = lane & 15;
#pragma unroll
        for (int t = 0; t < 3; ++t) {
            int n0 = (wave + t * 8) * 16;
#pragma unroll
            for (int i = 0; i < 8; ++i) {
                gi[half * 8 + i][n0 + col] = ci[t][i];
                gh[half * 8 + i][n0 + col] = cg[t][i];
            }
        }
    }
    if (threadIdx.x < 16) {
        int hn = 1;
        if (use_mask) {
            hn = 0;
            for (int k = 0; k < DEG; ++k)
                hn |= mask[(size_t)(m0 + threadIdx.x) * DEG + k];
        }
        hasnbr[threadIdx.x] = hn;
    }
    __syncthreads();

    // --- pointwise GRU phase ---
    for (int e = threadIdx.x; e < 16 * DDIM; e += 256) {
        int row = e >> 7, d = e & 127;
        size_t idx = (size_t)(m0 + row) * DDIM + d;
        float ir = gi[row][d]       + b_ih[d];
        float iz = gi[row][128 + d] + b_ih[128 + d];
        float in = gi[row][256 + d] + b_ih[256 + d];
        float hr = gh[row][d]       + b_hh[d];
        float hz = gh[row][128 + d] + b_hh[128 + d];
        float hn = gh[row][256 + d] + b_hh[256 + d];
        float r  = 1.f / (1.f + __expf(-(ir + hr)));
        float z  = 1.f / (1.f + __expf(-(iz + hz)));
        float nn = tanhf(in + r * hn);
        float hp = Hprev[idx];
        float h  = (1.f - z) * nn + z * hp;
        if (use_mask && !hasnbr[row]) h = hp;   // no in-neighbors: keep h1
        Hout[idx] = h;
        if (HoutB) HoutB[idx] = f2bf(h);
    }
}

// ---------------- generic GEMM + bias: C = A @ W^T + b ----------------
__global__ __launch_bounds__(256) void gemm_bias_kernel(
    const unsigned short* __restrict__ A,    // [M,128] bf16
    const unsigned short* __restrict__ W,    // [Ncols,128] bf16
    const float* __restrict__ bias,          // [Ncols]
    float* __restrict__ Cf,                  // nullable
    unsigned short* __restrict__ Cb,         // nullable
    int Ncols)
{
    const int m0   = blockIdx.x * 16;
    const int lane = threadIdx.x & 31;
    const int wave = threadIdx.x >> 5;
    const int n0   = (blockIdx.y * 8 + wave) * 16;
    if (n0 >= Ncols) return;

    v8f c = (v8f){};
#pragma unroll
    for (int k0 = 0; k0 < DDIM; k0 += 32) {
        v16bf a = load_frag_a(A, DDIM, m0, k0, lane);
        v16bf b = load_frag_b(W, DDIM, n0, k0, lane);
        c = WMMA_BF16(a, b, c);
    }
    const int half = lane >> 4, col = lane & 15;
    const float bv = bias[n0 + col];
#pragma unroll
    for (int i = 0; i < 8; ++i) {
        float v = c[i] + bv;
        size_t idx = (size_t)(m0 + half * 8 + i) * Ncols + n0 + col;
        if (Cf) Cf[idx] = v;
        if (Cb) Cb[idx] = f2bf(v);
    }
}

// ---------------- per-node multi-head attention over 16 neighbors ----------------
// one wave per (node, head); lane == element of head-dim (HD==32==wave32)
__global__ __launch_bounds__(256) void attn_kernel(
    const float* __restrict__ qkv,           // [N,384]: q|k|v
    const int* __restrict__ nbr_idx,         // [N,DEG]
    const unsigned char* __restrict__ mask,  // [N,DEG]
    unsigned short* __restrict__ attb)       // [N,128] bf16
{
    const int task = blockIdx.x * 8 + (threadIdx.x >> 5);
    const int lane = threadIdx.x & 31;
    const int node = task >> 2;
    const int head = task & 3;
    const size_t base = (size_t)node * DEG;

    const float qv = qkv[(size_t)node * TD + head * HD + lane];

    float sc[DEG];
    float smax = -1e30f;
#pragma unroll
    for (int k = 0; k < DEG; ++k) {
        int nb = nbr_idx[base + k];
        float kv = qkv[(size_t)nb * TD + DDIM + head * HD + lane];
        float p = qv * kv;
#pragma unroll
        for (int off = 16; off > 0; off >>= 1) p += __shfl_xor(p, off, 32);
        float s = p * 0.17677669529663687f;          // 1/sqrt(32)
        if (!mask[base + k]) s = -1e9f;
        sc[k] = s;
        smax = fmaxf(smax, s);
    }
    float denom = 0.f;
#pragma unroll
    for (int k = 0; k < DEG; ++k) { sc[k] = __expf(sc[k] - smax); denom += sc[k]; }
    const float inv = 1.f / denom;
    float acc = 0.f;
#pragma unroll
    for (int k = 0; k < DEG; ++k) {
        int nb = nbr_idx[base + k];
        acc += sc[k] * qkv[(size_t)nb * TD + 2 * DDIM + head * HD + lane];
    }
    attb[(size_t)node * DDIM + head * HD + lane] = f2bf(acc * inv);
}

// ---------------- launcher ----------------
extern "C" void kernel_launch(void* const* d_in, const int* in_sizes, int n_in,
                              void* d_out, int out_size, void* d_ws, size_t ws_size,
                              hipStream_t stream) {
    const float* x    = (const float*)d_in[0];
    const float* h0   = (const float*)d_in[1];
    const float* W_ih = (const float*)d_in[2];
    const float* W_hh = (const float*)d_in[3];
    const float* b_ih = (const float*)d_in[4];
    const float* b_hh = (const float*)d_in[5];
    const float* Wqkv = (const float*)d_in[6];
    const float* bqkv = (const float*)d_in[7];
    const float* Wo   = (const float*)d_in[8];
    const float* bo   = (const float*)d_in[9];
    const int*   nbr_idx  = (const int*)d_in[10];
    const unsigned char* nbr_mask = (const unsigned char*)d_in[11];
    float* out = (float*)d_out;

    // workspace layout (bytes)
    char* ws = (char*)d_ws;
    const size_t szW3  = (size_t)TD * DDIM * 2;     //  98304
    const size_t szWo  = (size_t)DDIM * DDIM * 2;   //  32768
    const size_t szNDb = (size_t)NNODES * DDIM * 2; //  16 MB
    const size_t szNDf = (size_t)NNODES * DDIM * 4; //  32 MB
    size_t off = 0;
    unsigned short* Wih_b  = (unsigned short*)(ws + off); off += szW3;
    unsigned short* Whh_b  = (unsigned short*)(ws + off); off += szW3;
    unsigned short* Wqkv_b = (unsigned short*)(ws + off); off += szW3;
    unsigned short* Wo_b   = (unsigned short*)(ws + off); off += szWo;
    off = (off + 511) & ~(size_t)511;
    unsigned short* xb   = (unsigned short*)(ws + off); off += szNDb;
    unsigned short* h0b  = (unsigned short*)(ws + off); off += szNDb;
    float*          h1f  = (float*)(ws + off);          off += szNDf;
    unsigned short* h1b  = (unsigned short*)(ws + off); off += szNDb;
    float*          qkv  = (float*)(ws + off);          off += (size_t)NNODES * TD * 4;
    unsigned short* attb = (unsigned short*)(ws + off); off += szNDb;
    unsigned short* aob  = (unsigned short*)(ws + off); off += szNDb;
    (void)ws_size; (void)n_in; (void)in_sizes; (void)out_size;

    // 1) convert weights + activations to bf16
    auto cvt = [&](const float* s, unsigned short* d, int n) {
        cvt_f32_bf16_kernel<<<(n + 255) / 256, 256, 0, stream>>>(s, d, n);
    };
    cvt(W_ih, Wih_b,  TD * DDIM);
    cvt(W_hh, Whh_b,  TD * DDIM);
    cvt(Wqkv, Wqkv_b, TD * DDIM);
    cvt(Wo,   Wo_b,   DDIM * DDIM);
    cvt(x,  xb,  NNODES * DDIM);
    cvt(h0, h0b, NNODES * DDIM);

    const int mtiles = NNODES / 16;   // 4096

    // 2) h1 = GRU(x, h0)
    gru_fused_kernel<<<mtiles, 256, 0, stream>>>(
        xb, h0b, Wih_b, Whh_b, b_ih, b_hh, h0, h1f, h1b, nullptr, 0);

    // 3) qkv = h1 @ Wqkv^T + bqkv   (f32 for attention)
    gemm_bias_kernel<<<dim3(mtiles, 3), 256, 0, stream>>>(
        h1b, Wqkv_b, bqkv, qkv, nullptr, TD);

    // 4) attended (pre-projection), bf16
    attn_kernel<<<(NNODES * NHEAD) / 8, 256, 0, stream>>>(qkv, nbr_idx, nbr_mask, attb);

    // 5) out-proj: attended @ Wo^T + bo -> bf16
    gemm_bias_kernel<<<dim3(mtiles, 1), 256, 0, stream>>>(
        attb, Wo_b, bo, nullptr, aob, DDIM);

    // 6) h2 = GRU(attended, h1), masked fallback to h1 -> d_out (f32)
    gru_fused_kernel<<<mtiles, 256, 0, stream>>>(
        aob, h1b, Wih_b, Whh_b, b_ih, b_hh, h1f, out, nullptr, nbr_mask, 1);
}
